// CausalSelfAttention_14319420965174
// MI455X (gfx1250) — matrix operand order repaired and checked by reading, not compiled, not run on hardware
//
#include <hip/hip_runtime.h>
#include <hip/hip_bf16.h>
#include <math.h>

#define BB 8
#define TT 1024
#define CC 1024
#define HH 16
#define HD 64
#define C3 3072

typedef __attribute__((ext_vector_type(16))) _Float16 v16h;
typedef __attribute__((ext_vector_type(8)))  _Float16 v8h;
typedef __attribute__((ext_vector_type(8)))  float    v8f;
typedef __attribute__((ext_vector_type(4)))  unsigned int v4u;
typedef __attribute__((ext_vector_type(8)))  int v8i;
typedef __attribute__((ext_vector_type(4)))  int v4i;

union V16 { v16h v; v8h h[2]; };

__device__ __forceinline__ v8f wmma16x16x32(v16h a, v16h b, v8f c) {
  // D = A(16x32 f16) * B(32x16 f16) + C(16x16 f32)
  return __builtin_amdgcn_wmma_f32_16x16x32_f16(false, a, false, b, (short)0, c, false, false);
}

// ---------------------------------------------------------------------------
// Tensor Data Mover: 2-D tile load Global->LDS with per-row LDS padding.
// tile: tile_k halfs per row x tile_rows rows; global row stride in halfs.
// LDS layout produced: 64B of data then 16B pad per row (row stride 80B),
// via pad_interval=3 (16 DWORDs) / pad_amount=3 (4 DWORDs).
// Tensor dims set to 2^20 so OOB clamping never triggers (tile in-bounds).
// ---------------------------------------------------------------------------
__device__ __forceinline__ void tdm_load_2d(unsigned lds_addr, const void* gsrc,
                                            unsigned tile_k, unsigned tile_rows,
                                            unsigned row_stride_elems) {
  unsigned long long ga = (unsigned long long)(uintptr_t)gsrc;
  v4u g0;
  g0[0] = 1u;                                             // count=1, user D#
  g0[1] = lds_addr;                                       // LDS byte address
  g0[2] = (unsigned)ga;                                   // global_addr[31:0]
  g0[3] = (unsigned)((ga >> 32) & 0x1FFFFFFu) | (2u << 30);  // addr[56:32]|type=2
  v8i g1;
  g1[0] = (int)((1u << 16) | (1u << 20) | (3u << 22) | (3u << 25));
          // data_size=2B | pad_enable | pad_interval=16DW | pad_amount=4DW
  g1[1] = 0;                                              // barrier=0, dim0 lo16=0
  g1[2] = (int)(0x100000u >> 16);                         // dim0 hi16 (2^20), dim1 lo16=0
  g1[3] = (int)((0x100000u >> 16) | (tile_k << 16));      // dim1 hi16 | tile_dim0
  g1[4] = (int)tile_rows;                                 // tile_dim1 (tile_dim2=0)
  g1[5] = (int)row_stride_elems;                          // tensor_dim0_stride lo32
  g1[6] = 0;                                              // stride0 hi | stride1 lo
  g1[7] = 0;
  v4i z4 = {0, 0, 0, 0};
  v8i z8 = {0, 0, 0, 0, 0, 0, 0, 0};
  // 6-arg form (clang-23 / therock-10.0 headers): extra int32x8 before cpol.
  __builtin_amdgcn_tensor_load_to_lds(g0, g1, z4, z4, z8, 0);
}

// Reductions across the 16 lanes of a wave32 half (masks 1..8 keep bit4 fixed).
__device__ __forceinline__ float half_max16(float v) {
#pragma unroll
  for (int m = 1; m < 16; m <<= 1) v = fmaxf(v, __shfl_xor(v, m, 32));
  return v;
}
__device__ __forceinline__ float half_sum16(float v) {
#pragma unroll
  for (int m = 1; m < 16; m <<= 1) v += __shfl_xor(v, m, 32);
  return v;
}

// ---------------- prep kernels ----------------
__global__ void cvt_f32_to_f16(const float* __restrict__ src,
                               _Float16* __restrict__ dst, int n) {
  int i = blockIdx.x * blockDim.x + threadIdx.x;
  int stride = gridDim.x * blockDim.x;
  for (; i < n; i += stride) dst[i] = (_Float16)src[i];
}

// src [K][N] f32 (row-major) -> dst [N][K] f16 (row-major), i.e. transpose.
__global__ void transpose_to_f16(const float* __restrict__ src,
                                 _Float16* __restrict__ dst, int K, int N) {
  int i = blockIdx.x * blockDim.x + threadIdx.x;
  int total = K * N;
  int stride = gridDim.x * blockDim.x;
  for (; i < total; i += stride) {
    int n = i / K, k = i - n * K;
    dst[i] = (_Float16)src[(size_t)k * N + n];
  }
}

// ============================================================================
// Tiled f16 GEMM core: block = 8 waves = 128 rows x 128 cols of the output.
// B k-slice (128 cols x 32 k, [col][k] layout) is double-buffered in LDS.
// Each wave TDM-loads its own 16-row slice of the next stage (TENSORcnt),
// overlapping the copy with the 8 WMMAs on the current stage. One barrier per
// k-step publishes the arrived stage AND proves all waves finished reading
// the other stage.
// acc[t] element (r,lane): row m0+r+8*(lane>>4), col n0+16*t+(lane&15).
// ============================================================================
#define BCOLS 128
#define BPAD  40  // padded row stride in halfs (80 B), matches TDM pad config

__device__ __forceinline__ void gemm_tile_128x128(
    const _Float16* __restrict__ A, const _Float16* __restrict__ wT,
    _Float16 (*Bs)[BCOLS][BPAD],  // [2][128][40]
    int m_blk, int n0, v8f acc[8]) {
  const int tid  = threadIdx.x;
  const int lane = tid & 31;
  const int wave = tid >> 5;
  const int half = lane >> 4, l16 = lane & 15;
  const int m0 = m_blk + wave * 16;

  // per-wave TDM slice: 16 consecutive B columns (= tile rows), 32 k per row
  const _Float16* gslice = wT + (size_t)(n0 + wave * 16) * CC;
  const unsigned ldsS0 = (unsigned)(uintptr_t)&Bs[0][wave * 16][0];
  const unsigned ldsS1 = (unsigned)(uintptr_t)&Bs[1][wave * 16][0];

  const _Float16* arow = A + (size_t)(m0 + l16) * CC + 8 * half;

  // prologue: stage 0 <- k-slice 0
  tdm_load_2d(ldsS0, gslice, 32, 16, CC);

  for (int i = 0; i < CC / 32; ++i) {
    const int k0 = i * 32;
    const int cur = i & 1;
    // A fragment (registers, private to this wave)
    V16 a;
    a.h[0] = *(const v8h*)(arow + k0);
    a.h[1] = *(const v8h*)(arow + k0 + 16);
    __builtin_prefetch(arow + k0 + 256, 0, 1);
    __builtin_amdgcn_s_wait_tensorcnt(0);  // my slice of stage cur landed
    __syncthreads();   // everyone's slice landed; everyone done with cur^1
    if (k0 + 32 < CC)  // overlap next TDM copy with this stage's WMMAs
      tdm_load_2d(cur ? ldsS0 : ldsS1, gslice + k0 + 32, 32, 16, CC);
#pragma unroll
    for (int t = 0; t < 8; ++t) {
      V16 b;
      b.h[0] = *(const v8h*)(&Bs[cur][16 * t + l16][8 * half]);
      b.h[1] = *(const v8h*)(&Bs[cur][16 * t + l16][8 * half + 16]);
      acc[t] = wmma16x16x32(a.v, b.v, acc[t]);
    }
  }
}

// ---------------- QKV GEMM: qkv = x @ w_attn + b_attn ----------------
// Scatters q -> [BH][T][HD], k -> [BH][T][HD], v -> transposed [BH][HD][T].
__global__ void __launch_bounds__(256) qkv_gemm(
    const _Float16* __restrict__ x16, const _Float16* __restrict__ wT,
    const float* __restrict__ bias, _Float16* __restrict__ qb,
    _Float16* __restrict__ kb, _Float16* __restrict__ vT) {
  __shared__ __align__(16) _Float16 Bs[2][BCOLS][BPAD];
  const int lane = threadIdx.x & 31;
  const int wave = threadIdx.x >> 5;
  const int half = lane >> 4, l16 = lane & 15;
  const int NB = C3 / BCOLS;  // 24 column blocks
  int bm = blockIdx.x / NB, bn = blockIdx.x % NB;
  int m_blk = bm * 128, n0 = bn * BCOLS;

  v8f acc[8] = {};
  gemm_tile_128x128(x16, wT, Bs, m_blk, n0, acc);

  int m0 = m_blk + wave * 16;
#pragma unroll
  for (int t = 0; t < 8; ++t) {
    int c = n0 + 16 * t + l16;
    float bv = bias[c];
    int which = c >> 10;  // 0=q 1=k 2=v (uniform across a 128-col block)
    int cc = c & 1023;
    int h = cc >> 6, d = cc & 63;
#pragma unroll
    for (int r = 0; r < 8; ++r) {
      int m = m0 + r + 8 * half;
      int b_ = m >> 10, tok = m & 1023;
      size_t bh = (size_t)(b_ * HH + h);
      _Float16 val = (_Float16)(acc[t][r] + bv);
      if (which == 0)      qb[(bh * TT + tok) * HD + d] = val;
      else if (which == 1) kb[(bh * TT + tok) * HD + d] = val;
      else                 vT[(bh * HD + d) * TT + tok] = val;
    }
  }
}

// ---------------- output projection: out = y @ w_proj + b_proj (f32) -------
__global__ void __launch_bounds__(256) proj_gemm(
    const _Float16* __restrict__ y16, const _Float16* __restrict__ wT,
    const float* __restrict__ bias, float* __restrict__ out) {
  __shared__ __align__(16) _Float16 Bs[2][BCOLS][BPAD];
  const int lane = threadIdx.x & 31;
  const int wave = threadIdx.x >> 5;
  const int half = lane >> 4, l16 = lane & 15;
  const int NB = CC / BCOLS;  // 8 column blocks
  int bm = blockIdx.x / NB, bn = blockIdx.x % NB;
  int m_blk = bm * 128, n0 = bn * BCOLS;

  v8f acc[8] = {};
  gemm_tile_128x128(y16, wT, Bs, m_blk, n0, acc);

  int m0 = m_blk + wave * 16;
#pragma unroll
  for (int t = 0; t < 8; ++t) {
    float bv = bias[n0 + 16 * t + l16];
#pragma unroll
    for (int r = 0; r < 8; ++r) {
      int m = m0 + r + 8 * half;
      out[(size_t)m * CC + n0 + 16 * t + l16] = acc[t][r] + bv;
    }
  }
}

// ---------------- causal attention (flash-style, one wave per 16 queries) ---
__global__ void __launch_bounds__(256) attn_kernel(
    const _Float16* __restrict__ qb, const _Float16* __restrict__ kb,
    const _Float16* __restrict__ vT, _Float16* __restrict__ y16) {
  __shared__ __align__(16) _Float16 Pt[8][16][40];  // per-wave 16x32 P tile
  const int lane = threadIdx.x & 31;
  const int wave = threadIdx.x >> 5;
  const int half = lane >> 4, l16 = lane & 15;
  int wtile = blockIdx.x * 8 + wave;
  int qt = wtile & 63;   // T/16 = 64 query tiles
  int bh = wtile >> 6;   // 0..127 (b*H + h)
  int m0 = qt * 16;

  // Q A-fragments for hd chunks [0,32) and [32,64), resident all loop.
  const _Float16* qrow = qb + ((size_t)bh * TT + (m0 + l16)) * HD;
  V16 aq0, aq1;
  aq0.h[0] = *(const v8h*)(qrow + 16 * half);
  aq0.h[1] = *(const v8h*)(qrow + 16 * half + 8);
  aq1.h[0] = *(const v8h*)(qrow + 32 + 16 * half);
  aq1.h[1] = *(const v8h*)(qrow + 32 + 16 * half + 8);

  float mrow[8], lsum[8];
  v8f o0 = {}, o1 = {}, o2 = {}, o3 = {};
#pragma unroll
  for (int r = 0; r < 8; ++r) { mrow[r] = -3.0e38f; lsum[r] = 0.0f; }

  const float sc = 0.125f;  // 1/sqrt(64)

  for (int nk0 = 0; nk0 < m0 + 16; nk0 += 32) {
    // S = Q K^T for keys [nk0, nk0+32): two 16-col fragments, K split 2x32.
    v8f s[2];
#pragma unroll
    for (int j = 0; j < 2; ++j) {
      int key = nk0 + 16 * j + l16;
      const _Float16* krow = kb + ((size_t)bh * TT + key) * HD + 16 * half;
      v16h b0 = *(const v16h*)(krow);        // hd 0..31 chunk
      v16h b1 = *(const v16h*)(krow + 32);   // hd 32..63 chunk
      v8f t = {};
      t = wmma16x16x32(aq0.v, b0, t);
      t = wmma16x16x32(aq1.v, b1, t);
      s[j] = t;
    }
    // online softmax (row stats per lane for its half's 8 rows)
#pragma unroll
    for (int r = 0; r < 8; ++r) {
      int qg = m0 + r + 8 * half;  // global query row
      float s0 = s[0][r] * sc;
      float s1 = s[1][r] * sc;
      if (nk0 + l16 > qg)      s0 = -INFINITY;
      if (nk0 + 16 + l16 > qg) s1 = -INFINITY;
      float nm = half_max16(fmaxf(s0, s1));
      nm = fmaxf(nm, mrow[r]);
      float rescale = __expf(mrow[r] - nm);
      mrow[r] = nm;
      float p0 = __expf(s0 - nm);
      float p1 = __expf(s1 - nm);
      lsum[r] = lsum[r] * rescale + half_sum16(p0 + p1);
      o0[r] *= rescale; o1[r] *= rescale; o2[r] *= rescale; o3[r] *= rescale;
      Pt[wave][r + 8 * half][l16]      = (_Float16)p0;
      Pt[wave][r + 8 * half][16 + l16] = (_Float16)p1;
    }
    // Re-stripe P (C/D layout) into an A fragment via per-wave LDS tile.
    V16 ap;
    ap.h[0] = *(const v8h*)(&Pt[wave][l16][16 * half]);
    ap.h[1] = *(const v8h*)(&Pt[wave][l16][16 * half + 8]);
    // O += P @ V  (K = 32 keys; vT is [hd][T] so B fragments are contiguous)
#pragma unroll
    for (int t = 0; t < 4; ++t) {
      int d = 16 * t + l16;
      v16h bv = *(const v16h*)(vT + ((size_t)bh * HD + d) * TT + nk0 + 16 * half);
      v8f* op = (t == 0) ? &o0 : (t == 1) ? &o1 : (t == 2) ? &o2 : &o3;
      *op = wmma16x16x32(ap.v, bv, *op);
    }
  }

  // normalize and write y back in [B,T,C] layout (f16) for the proj GEMM
  int b_ = bh >> 4, h = bh & 15;
#pragma unroll
  for (int r = 0; r < 8; ++r) {
    int tok = b_ * TT + m0 + r + 8 * half;
    float inv = 1.0f / lsum[r];
    size_t base = (size_t)tok * CC + h * HD;
    y16[base + 0 * 16 + l16] = (_Float16)(o0[r] * inv);
    y16[base + 1 * 16 + l16] = (_Float16)(o1[r] * inv);
    y16[base + 2 * 16 + l16] = (_Float16)(o2[r] * inv);
    y16[base + 3 * 16 + l16] = (_Float16)(o3[r] * inv);
  }
}

extern "C" void kernel_launch(void* const* d_in, const int* in_sizes, int n_in,
                              void* d_out, int out_size, void* d_ws,
                              size_t ws_size, hipStream_t stream) {
  const float* x      = (const float*)d_in[0];
  const float* w_attn = (const float*)d_in[1];
  const float* b_attn = (const float*)d_in[2];
  const float* w_proj = (const float*)d_in[3];
  const float* b_proj = (const float*)d_in[4];
  float* out = (float*)d_out;

  char* ws = (char*)d_ws;
  size_t off = 0;
  _Float16* x16 = (_Float16*)(ws + off); off += (size_t)BB * TT * CC * 2;      // 16 MB
  _Float16* wTa = (_Float16*)(ws + off); off += (size_t)C3 * CC * 2;           //  6 MB
  _Float16* wTp = (_Float16*)(ws + off); off += (size_t)CC * CC * 2;           //  2 MB
  _Float16* qb  = (_Float16*)(ws + off); off += (size_t)BB * HH * TT * HD * 2; // 16 MB
  _Float16* kb  = (_Float16*)(ws + off); off += (size_t)BB * HH * TT * HD * 2; // 16 MB
  _Float16* vT  = (_Float16*)(ws + off); off += (size_t)BB * HH * TT * HD * 2; // 16 MB
  _Float16* y16 = (_Float16*)(ws + off); off += (size_t)BB * TT * CC * 2;      // 16 MB

  cvt_f32_to_f16<<<2048, 256, 0, stream>>>(x, x16, BB * TT * CC);
  transpose_to_f16<<<2048, 256, 0, stream>>>(w_attn, wTa, CC, C3);
  transpose_to_f16<<<1024, 256, 0, stream>>>(w_proj, wTp, CC, CC);

  // 64 row blocks x 24 col blocks (128x128 tile per block)
  qkv_gemm<<<64 * 24, 256, 0, stream>>>(x16, wTa, b_attn, qb, kb, vT);
  // 128 (b,h) x 64 query tiles, 8 waves per block
  attn_kernel<<<(128 * 64) / 8, 256, 0, stream>>>(qb, kb, vT, y16);
  // 64 x 8 blocks
  proj_gemm<<<64 * 8, 256, 0, stream>>>(y16, wTp, b_proj, out);
}
